// H_O_EdgeApplyMoudle_712964571355
// MI455X (gfx1250) — matrix operand order repaired and checked by reading, not compiled, not run on hardware
//
#include <hip/hip_runtime.h>

typedef __attribute__((ext_vector_type(16))) _Float16 v16h;
typedef __attribute__((ext_vector_type(8)))  _Float16 v8h;
typedef __attribute__((ext_vector_type(4)))  _Float16 v4h;
typedef __attribute__((ext_vector_type(8))) float     v8f;
typedef __attribute__((ext_vector_type(4))) float     v4f;

#define D_IN   256   // 2*d (concat src|dst)
#define D_OUT  128   // out features
#define WAVES  8     // 8 wave32 per block
#define MTILES 2     // M-tiles (16 edges each) per wave -> 256 edges per block

// ---- one-time n_f f32 -> f16 conversion into workspace ----
__global__ __launch_bounds__(256)
void nf_to_f16(const float* __restrict__ in, _Float16* __restrict__ out,
               long long n4) {
    long long p = (long long)blockIdx.x * 256 + threadIdx.x;   // float4 index
    if (p < n4) {
        v4f w = ((const v4f*)in)[p];
        v4h h;
        h[0] = (_Float16)w[0]; h[1] = (_Float16)w[1];
        h[2] = (_Float16)w[2]; h[3] = (_Float16)w[3];
        ((v4h*)out)[p] = h;
    }
}

// Each wave owns 2 M-tiles (32 edges) and sweeps all 8 N-tiles; every B
// fragment loaded from LDS feeds TWO independent wmmas (half the LDS traffic
// per FLOP, two independent accumulation chains).
// 8 nt x 8 kk x 2 mt => 128 v_wmma_f32_16x16x32_f16 per wave, fully unrolled.
template<bool F16IN>
__global__ __launch_bounds__(256)
void gat_edge_kernel(const void*  __restrict__ nf,        // f16 or f32 rows [*,128]
                     const int*   __restrict__ src_idx,
                     const int*   __restrict__ dst_idx,
                     const float* __restrict__ W_edge,    // [128, 256] row-major
                     const float* __restrict__ W_attn,    // [1, 128]
                     float* __restrict__ e_out,           // [E, 128]
                     float* __restrict__ a_out,           // [E]
                     long long E)
{
    // W_edge as f16, row-major [n][k], XOR-swizzled at 16B-chunk granularity:
    // chunk' = chunk ^ (n & 15)  -> conflict-free ds_load_b128 for B fragments.
    __shared__ __align__(16) _Float16 Wlds[D_OUT * D_IN];   // exactly 64 KB

    const int t = threadIdx.x;

    // ---- stage W_edge (f32 -> f16) into swizzled LDS: 8192 float4, 32/thread ----
    {
        const v4f* Wv = (const v4f*)W_edge;
        #pragma unroll
        for (int i = 0; i < 32; ++i) {
            int p  = t + i * 256;          // float4 index
            v4f w  = Wv[p];
            int gk = p << 2;               // flat f32 index
            int n  = gk >> 8;              // row (out feature), 0..127
            int k  = gk & 255;             // col (in feature),  0..255 (mult of 4)
            int chunk = k >> 3;            // 16B chunk within row, 0..31
            int addr  = n * 256 + ((chunk ^ (n & 15)) << 3) + (k & 7);
            v4h h;
            h[0] = (_Float16)w[0]; h[1] = (_Float16)w[1];
            h[2] = (_Float16)w[2]; h[3] = (_Float16)w[3];
            *(v4h*)(&Wlds[addr]) = h;
        }
    }
    __syncthreads();

    const int lane = t & 31;
    const int lm   = lane & 15;     // N column / M row within tile
    const int hi   = lane >> 4;     // half-wave select
    const int wave = t >> 5;

    const long long mtile0 = ((long long)blockIdx.x * WAVES + wave) * MTILES;
    if (mtile0 * 16 >= E) return;   // wave-uniform; no barriers after this

    // ---- gather A fragments for both M-tiles ----
    // A 16x32 f16 layout: lane = M; low lanes K {0-7,16-23}, high lanes {8-15,24-31}
    v16h afrag[MTILES][8];
    #pragma unroll
    for (int mt = 0; mt < MTILES; ++mt) {
        long long edge = (mtile0 + mt) * 16 + lm;
        long long ge   = (edge < E) ? edge : (E - 1);   // clamp for safe gather
        const long long si = src_idx[ge];
        const long long di = dst_idx[ge];
        if constexpr (F16IN) {
            const _Float16* srow = (const _Float16*)nf + si * 128;
            const _Float16* drow = (const _Float16*)nf + di * 128;
            #pragma unroll
            for (int kk = 0; kk < 8; ++kk) {
                const _Float16* row = (kk < 4) ? srow : drow; // K<128 src else dst
                int rk = (kk & 3) * 32 + hi * 8;
                v8h x = *(const v8h*)(row + rk);        // 16B contiguous
                v8h y = *(const v8h*)(row + rk + 16);   // 16B contiguous
                v16h a;
                #pragma unroll
                for (int j = 0; j < 8; ++j) { a[j] = x[j]; a[8 + j] = y[j]; }
                afrag[mt][kk] = a;
            }
        } else {
            const float* srow = (const float*)nf + si * 128;
            const float* drow = (const float*)nf + di * 128;
            #pragma unroll
            for (int kk = 0; kk < 8; ++kk) {
                const float* row = (kk < 4) ? srow : drow;
                int rk = (kk & 3) * 32 + hi * 8;
                v4f x0 = *(const v4f*)(row + rk);
                v4f x1 = *(const v4f*)(row + rk + 4);
                v4f y0 = *(const v4f*)(row + rk + 16);
                v4f y1 = *(const v4f*)(row + rk + 20);
                v16h a;
                #pragma unroll
                for (int j = 0; j < 4; ++j) {
                    a[j]      = (_Float16)x0[j];
                    a[4 + j]  = (_Float16)x1[j];
                    a[8 + j]  = (_Float16)y0[j];
                    a[12 + j] = (_Float16)y1[j];
                }
                afrag[mt][kk] = a;
            }
        }
    }

    float pA[MTILES][8];
    #pragma unroll
    for (int mt = 0; mt < MTILES; ++mt)
        #pragma unroll
        for (int g = 0; g < 8; ++g) pA[mt][g] = 0.f;

    // ---- GEMM: 8 N-tiles x 8 K-steps, each B fragment feeds MTILES wmmas ----
    #pragma unroll
    for (int nt = 0; nt < 8; ++nt) {
        const int n = nt * 16 + lm;       // B column == W_edge row (out feature)
        v8f acc[MTILES];
        #pragma unroll
        for (int mt = 0; mt < MTILES; ++mt) acc[mt] = (v8f){};
        #pragma unroll
        for (int kk = 0; kk < 8; ++kk) {
            // B 32x16 f16 layout: lane = N; low lanes K {0-7,16-23}, high lanes {8-15,24-31}
            // => two 16B chunks of W_edge row n at kk*32 + hi*8 and kk*32 + hi*8 + 16
            int c0 = kk * 4 + hi;         // first 16B chunk id
            v8h b0 = *(const v8h*)(&Wlds[n * 256 + (((c0    ) ^ lm) << 3)]);
            v8h b1 = *(const v8h*)(&Wlds[n * 256 + (((c0 + 2) ^ lm) << 3)]);
            v16h b;
            #pragma unroll
            for (int j = 0; j < 8; ++j) { b[j] = b0[j]; b[8 + j] = b1[j]; }
            #pragma unroll
            for (int mt = 0; mt < MTILES; ++mt)
                acc[mt] = __builtin_amdgcn_wmma_f32_16x16x32_f16(
                              false, afrag[mt][kk], false, b, (short)0,
                              acc[mt], false, false);
        }
        // epilogue: relu, store e_feat, accumulate attention partial
        float wa = W_attn[n];
        #pragma unroll
        for (int mt = 0; mt < MTILES; ++mt) {
            #pragma unroll
            for (int g = 0; g < 8; ++g) {
                // C/D layout: VGPR g -> M = g (lanes 0-15) / g+8 (lanes 16-31)
                float r = acc[mt][g] > 0.f ? acc[mt][g] : 0.f;
                long long em = (mtile0 + mt) * 16 + g + 8 * hi;
                if (em < E)
                    __builtin_nontemporal_store(r, e_out + em * (long long)D_OUT + n);
                pA[mt][g] += r * wa;
            }
        }
    }

    // ---- a_fact: reduce over N (16-lane stripe per half-wave), leaky_relu ----
    #pragma unroll
    for (int mt = 0; mt < MTILES; ++mt) {
        #pragma unroll
        for (int g = 0; g < 8; ++g) {
            float s = pA[mt][g];
            s += __shfl_xor(s, 8, 16);
            s += __shfl_xor(s, 4, 16);
            s += __shfl_xor(s, 2, 16);
            s += __shfl_xor(s, 1, 16);
            if (lm == 0) {
                long long em = (mtile0 + mt) * 16 + g + 8 * hi;
                if (em < E)
                    __builtin_nontemporal_store(s > 0.f ? s : 0.01f * s, a_out + em);
            }
        }
    }
}

extern "C" void kernel_launch(void* const* d_in, const int* in_sizes, int n_in,
                              void* d_out, int out_size, void* d_ws, size_t ws_size,
                              hipStream_t stream) {
    const float* n_f    = (const float*)d_in[0];
    const int*   src    = (const int*)  d_in[1];
    const int*   dst    = (const int*)  d_in[2];
    const float* W_edge = (const float*)d_in[3];
    const float* W_attn = (const float*)d_in[4];
    const long long E   = in_sizes[1];
    const long long NF  = in_sizes[0];            // n_nodes * 128

    float* e_out = (float*)d_out;                 // [E,128] first
    float* a_out = e_out + E * 128LL;             // then [E,1]

    const int edgesPerBlock = WAVES * MTILES * 16;            // 256
    const int blocks = (int)((E + edgesPerBlock - 1) / edgesPerBlock);
    const size_t need = (size_t)NF * sizeof(_Float16);

    if (ws_size >= need) {
        // one-time f32->f16 node-feature conversion, then f16-gather GEMM
        long long n4 = NF / 4;
        nf_to_f16<<<(int)((n4 + 255) / 256), 256, 0, stream>>>(
            n_f, (_Float16*)d_ws, n4);
        gat_edge_kernel<true><<<blocks, 256, 0, stream>>>(
            d_ws, src, dst, W_edge, W_attn, e_out, a_out, E);
    } else {
        gat_edge_kernel<false><<<blocks, 256, 0, stream>>>(
            n_f, src, dst, W_edge, W_attn, e_out, a_out, E);
    }
}
